// SoftmaxAttention_31035433681706
// MI455X (gfx1250) — matrix operand order repaired and checked by reading, not compiled, not run on hardware
//
#include <hip/hip_runtime.h>

// Flash attention, B=2,H=16,Lq=Lk=2048,D=128 (fp32 in/out), gfx1250 wave32.
// Transposed formulation: S^T = K*Q^T, O^T = V^T*P^T; per-lane scalar softmax
// stats; P^T built in-register via SWAPX16 ds_swizzle. Fragment preloading
// into arrays so DS latency overlaps WMMA/VALU instead of serializing.

typedef __attribute__((ext_vector_type(16))) _Float16 v16h;
typedef __attribute__((ext_vector_type(8)))  _Float16 v8h;
typedef __attribute__((ext_vector_type(2)))  __fp16   fp16x2;
typedef __attribute__((ext_vector_type(8)))  float    v8f;
typedef __attribute__((ext_vector_type(8)))  int      i8v;
typedef __attribute__((ext_vector_type(4)))  int      i4v;

#define ATT_B  2
#define ATT_H  16
#define ATT_LQ 2048
#define ATT_LK 2048
#define ATT_D  128
// (1/sqrt(128)) * log2(e): scale + base-2 folded into Q once.
#define ATT_QSCALE 0.12751879570751127f

#define KSTR  136   // kTile row stride in halves (padded: conflict-free frags)
#define VSTR  40    // vT row stride in halves
#define OXSTR 132   // epilogue f32 row stride (16B-aligned rows, spread banks)
#define SWAPX16 0x401F  // ds_swizzle: xor=0x10, or=0, and=0x1f

#define WMMA_F16(A, Bm, C) \
  __builtin_amdgcn_wmma_f32_16x16x32_f16(false, (A), false, (Bm), (short)0, (C), false, false)

__device__ __forceinline__ float swap16f(float x) {
  return __int_as_float(__builtin_amdgcn_ds_swizzle(__float_as_int(x), SWAPX16));
}
__device__ __forceinline__ int pkh(float a, float b) {  // 2xf32 -> packed f16
  fp16x2 t = __builtin_amdgcn_cvt_pkrtz(a, b);
  return __builtin_bit_cast(int, t);
}
__device__ __forceinline__ v8h cvt8s(const float* p, float s) {
  float4 x0 = ((const float4*)p)[0];
  float4 x1 = ((const float4*)p)[1];
  v8h r;
  r[0] = (_Float16)(x0.x * s); r[1] = (_Float16)(x0.y * s);
  r[2] = (_Float16)(x0.z * s); r[3] = (_Float16)(x0.w * s);
  r[4] = (_Float16)(x1.x * s); r[5] = (_Float16)(x1.y * s);
  r[6] = (_Float16)(x1.z * s); r[7] = (_Float16)(x1.w * s);
  return r;
}
__device__ __forceinline__ v16h cat16(v8h lo, v8h hi) {
  return __builtin_shufflevector(lo, hi, 0,1,2,3,4,5,6,7,8,9,10,11,12,13,14,15);
}
__device__ __forceinline__ v16h ldfrag(const _Float16* p) {  // two b128 loads
  return cat16(*(const v8h*)p, *(const v8h*)(p + 16));
}

__global__ __launch_bounds__(256)
void fa_fwd_f16wmma(const float* __restrict__ q,
                    const float* __restrict__ k,
                    const float* __restrict__ v,
                    float* __restrict__ out) {
  // Loop phase: kTile [32][KSTR] halves (8704B) + vT [128][VSTR] (10240B).
  // Epilogue (after barrier): oX = 8 waves * 16*OXSTR f32 (67584B). Shared.
  __shared__ __align__(16) unsigned char smem[8 * 16 * OXSTR * 4];
  _Float16* kTile = (_Float16*)smem;
  _Float16* vT    = (_Float16*)(smem + 8704);

  const int tid  = threadIdx.x;
  const int bh   = blockIdx.y;
  const int wave = tid >> 5;
  const int lane = tid & 31;
  const int col  = lane & 15;   // this lane's query (N) / frag row (M)
  const int hi   = lane >> 4;   // lane half
  const int abase = hi * 8;
  const int q0   = blockIdx.x * 128 + wave * 16;

  // ---- Q^T as B-fragments (32feat x 16query), pre-scaled f16 -------------
  // B 32x16 layout: lane holds N=col, K = hi*8+e (e<8), 16+hi*8+(e-8) (e>=8).
  const float* qrow = q + ((size_t)bh * ATT_LQ + q0 + col) * ATT_D;
  v16h qb[4];
#pragma unroll
  for (int c = 0; c < 4; ++c) {
    const float* p = qrow + c * 32 + hi * 8;
    qb[c] = cat16(cvt8s(p, ATT_QSCALE), cvt8s(p + 16, ATT_QSCALE));
  }

  // ---- Per-lane (per-query) running state; O^T accumulators --------------
  float mstat = -1.0e30f, lstat = 0.0f;
  v8f ot[8];
#pragma unroll
  for (int t = 0; t < 8; ++t) ot[t] = (v8f){0.f,0.f,0.f,0.f,0.f,0.f,0.f,0.f};

  // staging maps: K: key=tid/8, 16-feat seg; V: key-pair = 2*(tid&15),
  // 8-feat seg = (tid>>4)*8 (adjacent keys -> adjacent halves in vT rows).
  const int kk = tid >> 3, kf = (tid & 7) << 4;
  const int vk2 = (tid & 15) << 1, vf8 = (tid >> 4) << 3;
  const float* kg = k + ((size_t)bh * ATT_LK + kk) * ATT_D + kf;
  const float* vg = v + ((size_t)bh * ATT_LK + vk2) * ATT_D + vf8;

  const int NT = ATT_LK / 32;
  for (int kt = 0; kt < NT; ++kt) {
    __syncthreads();
    // ---- Stage K (2x b128 stores) and V (pair-packed b32 stores) ---------
    {
      const float* kp = kg + (size_t)kt * 32 * ATT_D;
      const float* vp = vg + (size_t)kt * 32 * ATT_D;
      float4 a0 = ((const float4*)kp)[0], a1 = ((const float4*)kp)[1];
      float4 a2 = ((const float4*)kp)[2], a3 = ((const float4*)kp)[3];
      i4v w0 = (i4v){pkh(a0.x,a0.y), pkh(a0.z,a0.w), pkh(a1.x,a1.y), pkh(a1.z,a1.w)};
      i4v w1 = (i4v){pkh(a2.x,a2.y), pkh(a2.z,a2.w), pkh(a3.x,a3.y), pkh(a3.z,a3.w)};
      *(i4v*)&kTile[kk * KSTR + kf]     = w0;
      *(i4v*)&kTile[kk * KSTR + kf + 8] = w1;
      // V: rows vk2 and vk2+1, 8 feats; pack (key,key+1) pairs per feat
      float4 r00 = ((const float4*)vp)[0], r01 = ((const float4*)vp)[1];
      float4 r10 = ((const float4*)(vp + ATT_D))[0], r11 = ((const float4*)(vp + ATT_D))[1];
      *(int*)&vT[(vf8 + 0) * VSTR + vk2] = pkh(r00.x, r10.x);
      *(int*)&vT[(vf8 + 1) * VSTR + vk2] = pkh(r00.y, r10.y);
      *(int*)&vT[(vf8 + 2) * VSTR + vk2] = pkh(r00.z, r10.z);
      *(int*)&vT[(vf8 + 3) * VSTR + vk2] = pkh(r00.w, r10.w);
      *(int*)&vT[(vf8 + 4) * VSTR + vk2] = pkh(r01.x, r11.x);
      *(int*)&vT[(vf8 + 5) * VSTR + vk2] = pkh(r01.y, r11.y);
      *(int*)&vT[(vf8 + 6) * VSTR + vk2] = pkh(r01.z, r11.z);
      *(int*)&vT[(vf8 + 7) * VSTR + vk2] = pkh(r01.w, r11.w);
      if (kt + 1 < NT) {  // high locality: next tile is consumed by this thread
        __builtin_prefetch(kp + 32 * ATT_D, 0, 3);
        __builtin_prefetch(vp + 32 * ATT_D, 0, 3);
      }
    }
    __syncthreads();

    // ---- Preload all K A-fragments, then S^T = K * Q^T -------------------
    // A 16x32: lane row M=col(key), feats abase+{e,16+e-8}.
    v16h kA[8];
#pragma unroll
    for (int c = 0; c < 4; ++c) {
      kA[c]     = ldfrag(&kTile[col * KSTR + c * 32 + abase]);
      kA[4 + c] = ldfrag(&kTile[(16 + col) * KSTR + c * 32 + abase]);
    }
    v8f st0 = (v8f){0.f,0.f,0.f,0.f,0.f,0.f,0.f,0.f};
    v8f st1 = (v8f){0.f,0.f,0.f,0.f,0.f,0.f,0.f,0.f};
#pragma unroll
    for (int c = 0; c < 4; ++c) st0 = WMMA_F16(kA[c], qb[c], st0);
#pragma unroll
    for (int c = 0; c < 4; ++c) st1 = WMMA_F16(kA[4 + c], qb[c], st1);

    // ---- Preload V^T A-fragments (DS latency hides under softmax) --------
    v16h vA[8];
#pragma unroll
    for (int t = 0; t < 8; ++t)
      vA[t] = cat16(*(const v8h*)&vT[(t * 16 + col) * VSTR + hi * 16], *(const v8h*)&vT[(t * 16 + col) * VSTR + hi * 16 + 8]);

    // ---- Per-lane softmax over 32 keys (16 own + 16 in partner lane) -----
    float mx = fmaxf(st0[0], st1[0]);
#pragma unroll
    for (int r = 1; r < 8; ++r) mx = fmaxf(mx, fmaxf(st0[r], st1[r]));
    mx = fmaxf(mx, swap16f(mx));                 // combine with lane^16
    float mn   = fmaxf(mstat, mx);
    float corr = __builtin_amdgcn_exp2f(mstat - mn);
    float p0[8], p1[8], rs = 0.0f;
#pragma unroll
    for (int r = 0; r < 8; ++r) {
      p0[r] = __builtin_amdgcn_exp2f(st0[r] - mn);
      p1[r] = __builtin_amdgcn_exp2f(st1[r] - mn);
      rs += p0[r] + p1[r];
    }
    rs += swap16f(rs);                           // combine with lane^16
    lstat = lstat * corr + rs;
    mstat = mn;
    // Wave-uniform skip: rescale only if any lane's max advanced (corr<1).
    // Uniform branch keeps EXEC all-ones on both paths (WMMA requirement).
    if (__ballot(corr < 1.0f)) {
#pragma unroll
      for (int t = 0; t < 8; ++t)
#pragma unroll
        for (int r = 0; r < 8; ++r) ot[t][r] *= corr;  // scalar corr
    }

    // ---- Build P^T B-fragment (32key x 16query) in-register --------------
    // Lane needs keys hi*16+e. Own: sub0 keys hi*8+r, sub1 keys 16+hi*8+r.
    // Exchange the "inner" 8 values with lane^16 via SWAPX16.
    int s0p0 = pkh(p0[0], p0[1]), s0p1 = pkh(p0[2], p0[3]);
    int s0p2 = pkh(p0[4], p0[5]), s0p3 = pkh(p0[6], p0[7]);
    int s1p0 = pkh(p1[0], p1[1]), s1p1 = pkh(p1[2], p1[3]);
    int s1p2 = pkh(p1[4], p1[5]), s1p3 = pkh(p1[6], p1[7]);
    int g0 = __builtin_amdgcn_ds_swizzle(hi ? s0p0 : s1p0, SWAPX16);
    int g1 = __builtin_amdgcn_ds_swizzle(hi ? s0p1 : s1p1, SWAPX16);
    int g2 = __builtin_amdgcn_ds_swizzle(hi ? s0p2 : s1p2, SWAPX16);
    int g3 = __builtin_amdgcn_ds_swizzle(hi ? s0p3 : s1p3, SWAPX16);
    i8v pbv = (i8v){hi ? g0 : s0p0, hi ? g1 : s0p1, hi ? g2 : s0p2, hi ? g3 : s0p3,
                    hi ? s1p0 : g0, hi ? s1p1 : g1, hi ? s1p2 : g2, hi ? s1p3 : g3};
    v16h pB = __builtin_bit_cast(v16h, pbv);

    // ---- O^T += V^T * P^T : 8 feat tiles, K=32 keys each -----------------
#pragma unroll
    for (int t = 0; t < 8; ++t) ot[t] = WMMA_F16(vA[t], pB, ot[t]);
  }

  // ---- Epilogue: transpose O^T via LDS, store coalesced b128 -------------
  __syncthreads();  // all waves done with kTile/vT; reuse smem as f32 oX
  float inv = 1.0f / lstat;
  float* oX = (float*)smem + (size_t)wave * 16 * OXSTR;
#pragma unroll
  for (int t = 0; t < 8; ++t) {
    float4 lo4 = {ot[t][0] * inv, ot[t][1] * inv, ot[t][2] * inv, ot[t][3] * inv};
    float4 hi4 = {ot[t][4] * inv, ot[t][5] * inv, ot[t][6] * inv, ot[t][7] * inv};
    *(float4*)&oX[col * OXSTR + t * 16 + abase]     = lo4;
    *(float4*)&oX[col * OXSTR + t * 16 + abase + 4] = hi4;
  }
  asm volatile("s_wait_dscnt 0" ::: "memory");
  float* orow = out + ((size_t)bh * ATT_LQ + q0) * ATT_D;
#pragma unroll
  for (int qq = 0; qq < 16; ++qq) {
    float4 val = *(const float4*)&oX[qq * OXSTR + lane * 4];
    *(float4*)&orow[(size_t)qq * ATT_D + lane * 4] = val;
  }
}

extern "C" void kernel_launch(void* const* d_in, const int* in_sizes, int n_in,
                              void* d_out, int out_size, void* d_ws, size_t ws_size,
                              hipStream_t stream) {
  (void)in_sizes; (void)n_in; (void)out_size; (void)d_ws; (void)ws_size;
  const float* q = (const float*)d_in[0];
  const float* k = (const float*)d_in[1];
  const float* v = (const float*)d_in[2];
  float* out = (float*)d_out;

  dim3 grid(ATT_LQ / 128, ATT_B * ATT_H);  // (16, 32)
  dim3 block(256);                          // 8 waves, wave32
  fa_fwd_f16wmma<<<grid, block, 0, stream>>>(q, k, v, out);
}
